// FlashAttention_20641612824805
// MI455X (gfx1250) — compile-verified, matrix operand order checked by reading, not run on hardware
//
#include <hip/hip_runtime.h>

#define TDIM   2048
#define NHEADS 16
#define NKVH   8
#define HDIM   128
#define KC     128   // K-slab staged in LDS per GEMM stage

typedef __attribute__((ext_vector_type(16))) __bf16 v16bf;
typedef __attribute__((ext_vector_type(8)))  float  v8f;

typedef __attribute__((ext_vector_type(4))) unsigned int tdm_v4u;
typedef __attribute__((ext_vector_type(8))) int          tdm_v8i;
typedef __attribute__((ext_vector_type(4))) int          tdm_v4i;

union Frag16 { uint4 u[2]; v16bf v; };

__device__ __forceinline__ unsigned short bf16bits(float f) {
  union { float f; unsigned int u; } c; c.f = f;
  unsigned int u = c.u;
  u += 0x7fffu + ((u >> 16) & 1u);          // round-to-nearest-even
  return (unsigned short)(u >> 16);
}

// Load a 16x32 bf16 A-fragment (or column-major B-fragment) per the CDNA5
// 16-bit matrix VGPR layout: lane holds row (lane&15), half=lane>>4 selects
// K-subblocks; element e maps to k = 16*(e/8) + 8*half + e%8.
// Two contiguous 16B loads per lane (global_load_b128 or ds_load_b128).
__device__ __forceinline__ v16bf load_frag(const unsigned short* base, int ld,
                                           int row0, int k0, int lane) {
  const int row  = row0 + (lane & 15);
  const int half = lane >> 4;
  const unsigned short* p = base + (size_t)row * ld + k0 + half * 8;
  Frag16 f;
  f.u[0] = *(const uint4*)(p);
  f.u[1] = *(const uint4*)(p + 16);
  return f.v;
}

__device__ __forceinline__ v8f wmma_bf16(v16bf a, v16bf b, v8f c) {
  return __builtin_amdgcn_wmma_f32_16x16x32_bf16(false, a, false, b,
                                                 (short)0, c, false, false);
}

// ---------------------------------------------------------------------------
// Tensor Data Mover: DMA a 2-D bf16 tile (tile_d0 elems along K, tile_d1 rows)
// from a row-major tensor with row length tensor_d0 into contiguous LDS.
// D# packing per CDNA5 ISA (data_size=2 bytes,
// group0: count=1 | lds_addr | 57-bit global byte addr | type=2).
// This toolchain exposes the 6-arg builtin:
//   (uint32x4 g0, int32x8 g1, int32x4 g2, int32x4 g3, int32x8 g4, i32 cpol)
// groups 2..4 are unused for a 2-D tile (pass zeros).
__device__ __forceinline__ void tdm_load_2d(unsigned lds_off, const void* gptr,
                                            unsigned tensor_d0, unsigned tile_d0,
                                            unsigned tile_d1) {
  unsigned long long ga = (unsigned long long)gptr;
  unsigned long long s0 = tensor_d0;          // tensor_dim0_stride, data_size units
  tdm_v4u g0;
  g0[0] = 1u;                                                  // count=1
  g0[1] = lds_off;                                             // lds_addr (bytes)
  g0[2] = (unsigned)ga;                                        // global_addr[31:0]
  g0[3] = (unsigned)((ga >> 32) & 0x01ffffffu) | (2u << 30);   // ga[56:32] | type=2
  tdm_v8i g1;
  g1[0] = (int)(1u << 16);                                     // data_size=1 -> 2B
  g1[1] = (int)((tensor_d0 & 0xffffu) << 16);                  // tensor_dim0[15:0]
  g1[2] = (int)(((tensor_d0 >> 16) & 0xffffu) |
                ((tile_d1 & 0xffffu) << 16));                  // td0[31:16] | td1[15:0]
  g1[3] = (int)(((tile_d1 >> 16) & 0xffffu) |
                ((tile_d0 & 0xffffu) << 16));                  // td1[31:16] | tile_dim0
  g1[4] = (int)(tile_d1 & 0xffffu);                            // tile_dim1 (tile_dim2=0)
  g1[5] = (int)(unsigned)(s0 & 0xffffffffu);                   // dim0_stride[31:0]
  g1[6] = (int)(unsigned)((s0 >> 32) & 0xffffu);               // dim0_stride[47:32]
  g1[7] = 0;
  tdm_v4i gz4 = {0, 0, 0, 0};
  tdm_v8i gz8 = {0, 0, 0, 0, 0, 0, 0, 0};
  __builtin_amdgcn_tensor_load_to_lds(g0, g1, gz4, gz4, gz8, 0);
}

// ---------------------------------------------------------------------------
// Elementwise fp32 -> bf16
__global__ void f32_to_bf16_kernel(const float* __restrict__ in,
                                   unsigned short* __restrict__ out,
                                   long long n) {
  long long i = (long long)blockIdx.x * blockDim.x + threadIdx.x;
  if (i < n) out[i] = bf16bits(in[i]);
}

// W[k*N+n] (fp32, row-major KxN) -> Wt[n*K+k] (bf16, column-major)
__global__ void transpose_to_bf16_kernel(const float* __restrict__ W,
                                         unsigned short* __restrict__ Wt,
                                         int K, int N) {
  long long i = (long long)blockIdx.x * blockDim.x + threadIdx.x;
  if (i >= (long long)K * N) return;
  int k = (int)(i / N);
  int n = (int)(i % N);
  Wt[(size_t)n * K + k] = bf16bits(W[i]);
}

// ---------------------------------------------------------------------------
// C[MxN] = A[MxK] * Bt[NxK]^T, bf16 in / fp32 out.
// 8 waves (4x2) per block -> 64x32 C tile. K is consumed in KC=128 slabs that
// the TDM DMAs into LDS, double-buffered against the WMMA stream.
__global__ __launch_bounds__(256) void gemm_bf16_tdm_kernel(
    const unsigned short* __restrict__ A,
    const unsigned short* __restrict__ Bt,
    float* __restrict__ C, int M, int N, int K) {
  __shared__ unsigned short sA[2][64][KC];   // 2 x 16 KB
  __shared__ unsigned short sB[2][32][KC];   // 2 x  8 KB

  const int lane = threadIdx.x & 31;
  const int wv   = threadIdx.x >> 5;
  const int m0   = blockIdx.x * 64;
  const int n0   = blockIdx.y * 32;
  const int wm   = (wv & 3) * 16;
  const int wn   = (wv >> 2) * 16;
  const int nstages = K / KC;

  if (wv == 0) {   // TDM is a per-wave DMA instruction (EXEC ignored)
    tdm_load_2d((unsigned)(size_t)&sA[0][0][0], A  + (size_t)m0 * K, K, KC, 64);
    tdm_load_2d((unsigned)(size_t)&sB[0][0][0], Bt + (size_t)n0 * K, K, KC, 32);
    __builtin_amdgcn_s_wait_tensorcnt(0);
  }
  __syncthreads();

  v8f acc = {};
  for (int st = 0; st < nstages; ++st) {
    const int cur = st & 1;
    if (wv == 0 && st + 1 < nstages) {       // prefetch next slab into other buffer
      const size_t kk = (size_t)(st + 1) * KC;
      tdm_load_2d((unsigned)(size_t)&sA[cur ^ 1][0][0], A  + (size_t)m0 * K + kk, K, KC, 64);
      tdm_load_2d((unsigned)(size_t)&sB[cur ^ 1][0][0], Bt + (size_t)n0 * K + kk, K, KC, 32);
    }
#pragma unroll
    for (int ks = 0; ks < KC; ks += 32) {    // compute current slab from LDS
      v16bf a = load_frag(&sA[cur][0][0], KC, wm, ks, lane);
      v16bf b = load_frag(&sB[cur][0][0], KC, wn, ks, lane);
      acc = wmma_bf16(a, b, acc);
    }
    __syncthreads();                          // everyone done with cur buffer
    if (wv == 0) __builtin_amdgcn_s_wait_tensorcnt(0);
    __syncthreads();                          // next buffer visible to all waves
  }

  const int nn = n0 + wn + (lane & 15);
  const int mb = m0 + wm + (lane >> 4) * 8;  // C-fragment: VGPR r -> row r + 8*half
#pragma unroll
  for (int r = 0; r < 8; ++r)
    C[(size_t)(mb + r) * N + nn] = acc[r];
}

// ---------------------------------------------------------------------------
// RoPE + fp32->bf16, relayout [b*S+s, h, d] -> [b, h, s, d]
__global__ void rope_bf16_kernel(const float* __restrict__ src,
                                 unsigned short* __restrict__ dst,
                                 int nheads, int S, long long total) {
  long long idx = (long long)blockIdx.x * blockDim.x + threadIdx.x;
  if (idx >= total) return;                      // total = rows*nheads*64
  int d = (int)(idx % 64);
  int h = (int)((idx / 64) % nheads);
  long long row = idx / (64 * nheads);           // row = b*S + s
  int s = (int)(row % S);
  int b = (int)(row / S);
  float ang = (float)s * __expf(-(float)d * 0.14391156f);  // 10000^(-d/64)
  float cs = __cosf(ang), sn = __sinf(ang);
  const float* p = src + (size_t)row * nheads * HDIM + (size_t)h * HDIM;
  float x1 = p[d], x2 = p[d + 64];
  unsigned short* q = dst + ((size_t)(b * nheads + h) * S + s) * HDIM;
  q[d]      = bf16bits(x1 * cs - x2 * sn);
  q[d + 64] = bf16bits(x2 * cs + x1 * sn);
}

// V: [b*S+s, h, d] fp32 -> [b, h, d, s] bf16 (transposed, so PV B-fragments
// are contiguous along the key dimension)
__global__ void v_transpose_kernel(const float* __restrict__ src,
                                   unsigned short* __restrict__ dst,
                                   int S, long long total) {
  long long idx = (long long)blockIdx.x * blockDim.x + threadIdx.x;
  if (idx >= total) return;                      // total = rows*NKVH*HDIM
  int d = (int)(idx % HDIM);
  int h = (int)((idx / HDIM) % NKVH);
  long long row = idx / (HDIM * NKVH);
  int s = (int)(row % S);
  int b = (int)(row / S);
  float v = src[(size_t)row * NKVH * HDIM + (size_t)h * HDIM + d];
  dst[((size_t)(b * NKVH + h) * HDIM + d) * S + s] = bf16bits(v);
}

// ---------------------------------------------------------------------------
// Flash attention, causal, GQA (2 query heads per kv head).
// 8 waves/block; wave handles one 16-row query tile, loops 32-key chunks.
__global__ __launch_bounds__(256) void flash_attn_kernel(
    const unsigned short* __restrict__ qb,   // [B, H, S, D] bf16
    const unsigned short* __restrict__ kb,   // [B, KVH, S, D] bf16
    const unsigned short* __restrict__ vt,   // [B, KVH, D, S] bf16
    unsigned short* __restrict__ ob,         // [B*S, H*D] bf16
    int S) {
  __shared__ float          s_sc[8][16][32];
  __shared__ unsigned short s_p [8][16][32];
  __shared__ float          s_fac[8][16];
  __shared__ float          s_li [8][16];

  const int lane  = threadIdx.x & 31;
  const int wv    = threadIdx.x >> 5;
  const int tiles = S >> 7;                          // 128 query rows / block
  const int tile  = blockIdx.x % tiles;
  const int h     = (blockIdx.x / tiles) % NHEADS;
  const int b     = blockIdx.x / (tiles * NHEADS);
  const int q0    = tile * 128 + wv * 16;

  const unsigned short* qmat = qb + (size_t)(b * NHEADS + h)      * S * HDIM;
  const unsigned short* kmat = kb + (size_t)(b * NKVH + (h >> 1)) * S * HDIM;
  const unsigned short* vmat = vt + (size_t)(b * NKVH + (h >> 1)) * HDIM * S;

  v16bf qf[4];
#pragma unroll
  for (int i = 0; i < 4; ++i) qf[i] = load_frag(qmat, HDIM, q0, i * 32, lane);

  v8f o[8] = {};
  float mrun = -1e30f, lrun = 0.0f;                 // valid in lanes 0..15
  const float scale = 0.08838834764831845f;         // 1/sqrt(128)
  const int mloc = (lane >> 4) * 8;
  const int nn   = lane & 15;

  for (int kc = 0; kc < q0 + 16; kc += 32) {
    // ---- scores: q(16x128) x k^T(128x32) via 8 chained wmmas
    v8f s0 = {}, s1 = {};
#pragma unroll
    for (int i = 0; i < 4; ++i) {
      v16bf k0 = load_frag(kmat, HDIM, kc,      i * 32, lane);
      v16bf k1 = load_frag(kmat, HDIM, kc + 16, i * 32, lane);
      s0 = wmma_bf16(qf[i], k0, s0);
      s1 = wmma_bf16(qf[i], k1, s1);
    }
    // C-fragment -> row-major LDS
#pragma unroll
    for (int r = 0; r < 8; ++r) {
      s_sc[wv][mloc + r][nn]      = s0[r] * scale;
      s_sc[wv][mloc + r][nn + 16] = s1[r] * scale;
    }
    asm volatile("s_wait_dscnt 0" ::: "memory");    // cross-lane LDS hazard

    // ---- online softmax, one row per lane (lanes 0..15)
    if (lane < 16) {
      const int grow = q0 + lane;
      float mrow = mrun;
#pragma unroll
      for (int j = 0; j < 32; ++j) {
        float sv = ((kc + j) <= grow) ? s_sc[wv][lane][j] : -1e30f;
        s_sc[wv][lane][j] = sv;
        mrow = fmaxf(mrow, sv);
      }
      const float fac = __expf(mrun - mrow);
      float ps = 0.0f;
#pragma unroll
      for (int j = 0; j < 32; ++j) {
        float pv = __expf(s_sc[wv][lane][j] - mrow);
        ps += pv;
        s_p[wv][lane][j] = bf16bits(pv);
      }
      lrun = lrun * fac + ps;
      mrun = mrow;
      s_fac[wv][lane] = fac;
    }
    asm volatile("s_wait_dscnt 0" ::: "memory");

    // ---- rescale accumulators by per-row factors
    float fr[8];
#pragma unroll
    for (int r = 0; r < 8; ++r) fr[r] = s_fac[wv][mloc + r];
#pragma unroll
    for (int t = 0; t < 8; ++t)
#pragma unroll
      for (int r = 0; r < 8; ++r) o[t][r] *= fr[r];

    // ---- PV: P(16x32) x V(32x128): reload P as A-fragment from LDS
    v16bf pf = load_frag(&s_p[wv][0][0], 32, 0, 0, lane);
#pragma unroll
    for (int t = 0; t < 8; ++t) {
      v16bf vf = load_frag(vmat, S, t * 16, kc, lane);  // rows = head dims
      o[t] = wmma_bf16(pf, vf, o[t]);
    }
  }

  // ---- finalize: divide by row sums, store bf16 in [b*S+s, h*D+d]
  if (lane < 16) s_li[wv][lane] = 1.0f / lrun;
  asm volatile("s_wait_dscnt 0" ::: "memory");
  float li[8];
#pragma unroll
  for (int r = 0; r < 8; ++r) li[r] = s_li[wv][mloc + r];
#pragma unroll
  for (int t = 0; t < 8; ++t)
#pragma unroll
    for (int r = 0; r < 8; ++r)
      ob[((size_t)(b * S + q0 + mloc + r)) * (NHEADS * HDIM)
         + h * HDIM + t * 16 + nn] = bf16bits(o[t][r] * li[r]);
}

// ---------------------------------------------------------------------------
extern "C" void kernel_launch(void* const* d_in, const int* in_sizes, int n_in,
                              void* d_out, int out_size, void* d_ws, size_t ws_size,
                              hipStream_t stream) {
  const float* x  = (const float*)d_in[0];
  const float* Wq = (const float*)d_in[1];
  const float* Wk = (const float*)d_in[2];
  const float* Wv = (const float*)d_in[3];
  const float* Wo = (const float*)d_in[4];

  const int S    = 2048;
  const int rows = in_sizes[0] / TDIM;   // B*S
  const int B    = rows / S;
  const int NQ   = NHEADS * HDIM;        // 2048
  const int NKV  = NKVH * HDIM;          // 1024

  char* p = (char*)d_ws;
  auto take = [&](size_t bytes) -> char* {
    char* r = p; p += (bytes + 255) & ~(size_t)255; return r;
  };
  unsigned short* xb   = (unsigned short*)take((size_t)rows * TDIM * 2);
  unsigned short* WqT  = (unsigned short*)take((size_t)NQ  * TDIM * 2);
  unsigned short* WkT  = (unsigned short*)take((size_t)NKV * TDIM * 2);
  unsigned short* WvT  = (unsigned short*)take((size_t)NKV * TDIM * 2);
  unsigned short* WoT  = (unsigned short*)take((size_t)TDIM * NQ  * 2);
  unsigned short* qbf  = (unsigned short*)take((size_t)rows * NQ  * 2);
  unsigned short* kbf  = (unsigned short*)take((size_t)rows * NKV * 2);
  unsigned short* vtb  = (unsigned short*)take((size_t)rows * NKV * 2);
  float*          qf32 = (float*)take((size_t)rows * NQ  * 4);
  float*          kf32 = (float*)take((size_t)rows * NKV * 4);
  float*          vf32 = (float*)take((size_t)rows * NKV * 4);
  unsigned short* abf  = (unsigned short*)qf32;  // reuse: qf32 dead after RoPE

  const int T = 256;
  auto blocks = [&](long long n) { return (unsigned)((n + T - 1) / T); };

  // 1) activations -> bf16
  long long nx = (long long)rows * TDIM;
  f32_to_bf16_kernel<<<blocks(nx), T, 0, stream>>>(x, xb, nx);

  // 2) weights -> bf16, pre-transposed for contiguous B-fragment loads
  transpose_to_bf16_kernel<<<blocks((long long)TDIM * NQ),  T, 0, stream>>>(Wq, WqT, TDIM, NQ);
  transpose_to_bf16_kernel<<<blocks((long long)TDIM * NKV), T, 0, stream>>>(Wk, WkT, TDIM, NKV);
  transpose_to_bf16_kernel<<<blocks((long long)TDIM * NKV), T, 0, stream>>>(Wv, WvT, TDIM, NKV);
  transpose_to_bf16_kernel<<<blocks((long long)NQ  * TDIM), T, 0, stream>>>(Wo, WoT, NQ, TDIM);

  // 3) QKV projections (TDM double-buffered WMMA GEMMs, fp32 accumulate)
  gemm_bf16_tdm_kernel<<<dim3(rows / 64, NQ  / 32), T, 0, stream>>>(xb, WqT, qf32, rows, NQ,  TDIM);
  gemm_bf16_tdm_kernel<<<dim3(rows / 64, NKV / 32), T, 0, stream>>>(xb, WkT, kf32, rows, NKV, TDIM);
  gemm_bf16_tdm_kernel<<<dim3(rows / 64, NKV / 32), T, 0, stream>>>(xb, WvT, vf32, rows, NKV, TDIM);

  // 4) RoPE + bf16 relayout; V transpose
  long long nq = (long long)rows * NHEADS * 64;
  long long nk = (long long)rows * NKVH * 64;
  long long nv = (long long)rows * NKVH * HDIM;
  rope_bf16_kernel<<<blocks(nq), T, 0, stream>>>(qf32, qbf, NHEADS, S, nq);
  rope_bf16_kernel<<<blocks(nk), T, 0, stream>>>(kf32, kbf, NKVH,  S, nk);
  v_transpose_kernel<<<blocks(nv), T, 0, stream>>>(vf32, vtb, S, nv);

  // 5) Flash attention
  flash_attn_kernel<<<B * NHEADS * (S / 128), T, 0, stream>>>(qbf, kbf, vtb, abf, S);

  // 6) Output projection -> fp32 d_out
  gemm_bf16_tdm_kernel<<<dim3(rows / 64, TDIM / 32), T, 0, stream>>>(abf, WoT, (float*)d_out,
                                                                     rows, TDIM, NQ);
}
